// SparseCutAttention_26465588478185
// MI455X (gfx1250) — matrix operand order repaired and checked by reading, not compiled, not run on hardware
//
#include <hip/hip_runtime.h>
#include <math.h>

// Problem constants (from reference)
#define B_   2
#define C_   64
#define H_   8192
#define HB_  1024      // H / BLOCKS
#define NB_  7         // BLOCKS - 1
#define CUT_ 4
#define L_   4096      // CUT_ * HB_
#define TEMP_INV 1.4285714285714286f  // 1/0.7

typedef _Float16 h8   __attribute__((ext_vector_type(8)));
typedef _Float16 v16h __attribute__((ext_vector_type(16)));
typedef float    v8f  __attribute__((ext_vector_type(8)));

// ---------------------------------------------------------------------------
// Kernel 1: channel means of blockified q/k -> sq, sk [B,7,1024]; zero sums
// ---------------------------------------------------------------------------
__global__ void means_kernel(const float* __restrict__ q, const float* __restrict__ k,
                             float* __restrict__ sq, float* __restrict__ sk,
                             float* __restrict__ sums) {
  int t = blockIdx.x * blockDim.x + threadIdx.x;   // 2*7*1024 = 14336 threads
  if (t < 4) sums[t] = 0.f;
  int b   = t / (NB_ * HB_);
  int rem = t % (NB_ * HB_);
  int n   = rem / HB_;
  int h   = rem % HB_;
  const float* qb = q + (size_t)b * C_ * H_ + n * HB_ + h;
  const float* kb = k + (size_t)b * C_ * H_ + n * HB_ + h;
  float aq = 0.f, ak = 0.f;
  for (int c = 0; c < C_; ++c) {
    aq += qb[(size_t)c * H_];
    ak += kb[(size_t)c * H_];
  }
  sq[t] = aq * (1.f / C_);
  sk[t] = ak * (1.f / C_);
}

// ---------------------------------------------------------------------------
// Kernel 2: R = sq·skᵀ/8, gumbel, 8 Sinkhorn iters -> perm [B,7,7]
// ---------------------------------------------------------------------------
__global__ void perm_kernel(const float* __restrict__ sq, const float* __restrict__ sk,
                            const float* __restrict__ gu, float* __restrict__ perm) {
  int b   = blockIdx.x;
  int tid = threadIdx.x;          // 256
  __shared__ float red[256];
  __shared__ float Rm[49];
  const float* sqb = sq + b * NB_ * HB_;
  const float* skb = sk + b * NB_ * HB_;
  for (int p = 0; p < 49; ++p) {
    int i = p / 7, j = p % 7;
    float acc = 0.f;
    for (int h = tid; h < HB_; h += 256) acc += sqb[i * HB_ + h] * skb[j * HB_ + h];
    red[tid] = acc; __syncthreads();
    for (int off = 128; off > 0; off >>= 1) {
      if (tid < off) red[tid] += red[tid + off];
      __syncthreads();
    }
    if (tid == 0) Rm[p] = red[0] * 0.125f;   // * C^-0.5 = 1/8
    __syncthreads();
  }
  if (tid == 0) {
    float r[49];
    for (int p = 0; p < 49; ++p) {
      float u  = gu[b * 49 + p];
      float g  = -logf(-logf(u + 1e-6f) + 1e-6f);
      float Rv = Rm[p];
      float lg = (Rv > 0.f) ? logf(Rv) : -1e30f;   // log(relu(R))
      r[p] = (lg + g) * TEMP_INV;
    }
    for (int it = 0; it < 8; ++it) {
      for (int i = 0; i < 7; ++i) {           // logsumexp over axis=2 (cols)
        float m = r[i * 7];
        for (int j = 1; j < 7; ++j) m = fmaxf(m, r[i * 7 + j]);
        float s = 0.f;
        for (int j = 0; j < 7; ++j) s += __expf(r[i * 7 + j] - m);
        float lse = m + logf(s);
        for (int j = 0; j < 7; ++j) r[i * 7 + j] -= lse;
      }
      for (int j = 0; j < 7; ++j) {           // logsumexp over axis=1 (rows)
        float m = r[j];
        for (int i = 1; i < 7; ++i) m = fmaxf(m, r[i * 7 + j]);
        float s = 0.f;
        for (int i = 0; i < 7; ++i) s += __expf(r[i * 7 + j] - m);
        float lse = m + logf(s);
        for (int i = 0; i < 7; ++i) r[i * 7 + j] -= lse;
      }
    }
    for (int p = 0; p < 49; ++p) perm[b * 49 + p] = __expf(r[p]);
  }
}

// ---------------------------------------------------------------------------
// Kernel 3: block mix + cut -> ks f16 [B, L, C] row-major, vt f16 [B, C, L]
// ---------------------------------------------------------------------------
__global__ void sortcut_kernel(const float* __restrict__ k, const float* __restrict__ v,
                               const float* __restrict__ perm,
                               _Float16* __restrict__ ks, _Float16* __restrict__ vt) {
  int t   = blockIdx.x * 256 + threadIdx.x;    // 2*64*1024 = 131072
  int b   = t / (C_ * HB_);
  int rem = t % (C_ * HB_);
  int c   = rem / HB_;
  int h   = rem % HB_;
  const float* pp = perm + b * 49;
  const float* kb = k + (size_t)(b * C_ + c) * H_ + h;
  const float* vb = v + (size_t)(b * C_ + c) * H_ + h;
  float kv[7], vv[7];
  #pragma unroll
  for (int n = 0; n < 7; ++n) { kv[n] = kb[n * HB_]; vv[n] = vb[n * HB_]; }
  #pragma unroll
  for (int m = 0; m < CUT_; ++m) {
    float aks = 0.f, avs = 0.f;
    #pragma unroll
    for (int n = 0; n < 7; ++n) {
      float p = pp[n * 7 + m];
      aks += kv[n] * p;
      avs += vv[n] * p;
    }
    int l = m * HB_ + h;
    ks[((size_t)b * L_ + l) * C_ + c]        = (_Float16)aks;
    vt[((size_t)(b * C_ + c)) * L_ + l]      = (_Float16)avs;
  }
}

// ---------------------------------------------------------------------------
// Kernel 4: q transpose + fold 1/temperature -> qh f16 [B, H, C]
// ---------------------------------------------------------------------------
__global__ void qtrans_kernel(const float* __restrict__ q, _Float16* __restrict__ qh) {
  int t   = blockIdx.x * 256 + threadIdx.x;    // 2*64*8192, flat = (b*C+c)*H + hh
  int b   = t / (C_ * H_);
  int rem = t % (C_ * H_);
  int c   = rem / H_;
  int hh  = rem % H_;
  qh[((size_t)b * H_ + hh) * C_ + c] = (_Float16)(q[t] * TEMP_INV);
}

// ---------------------------------------------------------------------------
// Kernel 5: fused flash attention via v_wmma_f32_16x16x32_f16.
// Block = 256 threads (8 waves); each wave owns a 16-query strip.
// Key tiles of 128; S 16x128 in VGPRs; online softmax; P round-trips through
// a wave-private LDS tile to convert C/D layout -> A layout for P x V.
// ---------------------------------------------------------------------------
__launch_bounds__(256)
__global__ void attn_kernel(const _Float16* __restrict__ qh, const _Float16* __restrict__ ks,
                            const _Float16* __restrict__ vt, float* __restrict__ val,
                            float* __restrict__ sums) {
  __shared__ __align__(32) _Float16 pbuf[8][16 * 128];  // 32 KB, wave-private tiles
  __shared__ float redS[256];
  __shared__ float redS2[256];

  const int tid    = threadIdx.x;
  const int w      = tid >> 5;
  const int lane   = tid & 31;
  const int nlo    = lane & 15;
  const int hi     = lane >> 4;
  const int koff8  = hi * 8;
  const int koff16 = hi * 16;
  const int b      = blockIdx.y;
  const int strip  = blockIdx.x * 128 + w * 16;

  // A-fragments of Q strip (16x64 -> two 16x32 chunks), contiguous 16B loads
  const _Float16* qp = qh + ((size_t)b * H_ + strip + nlo) * C_;
  v16h aq0, aq1;
  {
    h8 l0 = *(const h8*)(qp + koff8);
    h8 h0 = *(const h8*)(qp + 16 + koff8);
    h8 l1 = *(const h8*)(qp + 32 + koff8);
    h8 h1 = *(const h8*)(qp + 48 + koff8);
    #pragma unroll
    for (int i = 0; i < 8; ++i) {
      aq0[i] = l0[i]; aq0[8 + i] = h0[i];
      aq1[i] = l1[i]; aq1[8 + i] = h1[i];
    }
  }

  const _Float16* ksb = ks + (size_t)b * L_ * C_;
  const _Float16* vtb = vt + (size_t)b * C_ * L_;

  v8f zero = {};
  v8f O[4] = {zero, zero, zero, zero};
  float m[8], lsum[8];
  #pragma unroll
  for (int r = 0; r < 8; ++r) { m[r] = -1e30f; lsum[r] = 0.f; }

  _Float16* pw = &pbuf[w][0];

  for (int kb0 = 0; kb0 < L_; kb0 += 128) {
    if (kb0 + 128 < L_) {   // hint next K/V tile into cache (global_prefetch_b8)
      __builtin_prefetch(ksb + (size_t)(kb0 + 128) * C_ + lane * 64, 0, 1);
      __builtin_prefetch(vtb + (size_t)nlo * L_ + kb0 + 128, 0, 1);
    }
    // ---- S = Q * K^T (scale folded into Q) ----
    v8f S[8];
    #pragma unroll
    for (int j = 0; j < 8; ++j) {
      const _Float16* kp = ksb + (size_t)(kb0 + j * 16 + nlo) * C_;
      v16h bk0 = __builtin_shufflevector(*(const h8*)(kp + koff8), *(const h8*)(kp + 16 + koff8), 0,1,2,3,4,5,6,7,8,9,10,11,12,13,14,15);
      v16h bk1 = __builtin_shufflevector(*(const h8*)(kp + 32 + koff8), *(const h8*)(kp + 48 + koff8), 0,1,2,3,4,5,6,7,8,9,10,11,12,13,14,15);
      v8f acc = zero;
      acc = __builtin_amdgcn_wmma_f32_16x16x32_f16(false, aq0, false, bk0, (short)0, acc, false, false);
      acc = __builtin_amdgcn_wmma_f32_16x16x32_f16(false, aq1, false, bk1, (short)0, acc, false, false);
      S[j] = acc;
    }
    // ---- online softmax (rows live within 16-lane halves) ----
    float mnew[8], scale[8];
    #pragma unroll
    for (int r = 0; r < 8; ++r) {
      float mj = S[0][r];
      #pragma unroll
      for (int j = 1; j < 8; ++j) mj = fmaxf(mj, S[j][r]);
      #pragma unroll
      for (int msk = 1; msk < 16; msk <<= 1) mj = fmaxf(mj, __shfl_xor(mj, msk, 32));
      mnew[r]  = fmaxf(m[r], mj);
      scale[r] = __expf(m[r] - mnew[r]);
      m[r]     = mnew[r];
    }
    #pragma unroll
    for (int ct = 0; ct < 4; ++ct)
      #pragma unroll
      for (int r = 0; r < 8; ++r) O[ct][r] *= scale[r];
    #pragma unroll
    for (int j = 0; j < 8; ++j)
      #pragma unroll
      for (int r = 0; r < 8; ++r) S[j][r] = __expf(S[j][r] - mnew[r]);
    #pragma unroll
    for (int r = 0; r < 8; ++r) {
      float s = 0.f;
      #pragma unroll
      for (int j = 0; j < 8; ++j) s += S[j][r];
      #pragma unroll
      for (int msk = 1; msk < 16; msk <<= 1) s += __shfl_xor(s, msk, 32);
      lsum[r] = lsum[r] * scale[r] + s;
    }
    // ---- P: C/D layout -> row-major LDS tile (16x128 f16) ----
    #pragma unroll
    for (int j = 0; j < 8; ++j)
      #pragma unroll
      for (int r = 0; r < 8; ++r)
        pw[(r + koff8) * 128 + j * 16 + nlo] = (_Float16)S[j][r];
    __syncthreads();   // LDS visibility (dscnt wait) before A-layout reads
    // ---- O += P * V (A from LDS, B contiguous from transposed V) ----
    #pragma unroll
    for (int ch = 0; ch < 4; ++ch) {
      const _Float16* pr = pw + nlo * 128 + ch * 32;
      h8 plo = *(const h8*)(pr + koff8);
      h8 phi = *(const h8*)(pr + 16 + koff8);
      v16h ap;
      #pragma unroll
      for (int i = 0; i < 8; ++i) { ap[i] = plo[i]; ap[8 + i] = phi[i]; }
      #pragma unroll
      for (int ct = 0; ct < 4; ++ct) {
        const _Float16* vp = vtb + (size_t)(ct * 16 + nlo) * L_ + kb0 + ch * 32 + koff8;
        v16h bv = __builtin_shufflevector(*(const h8*)vp, *(const h8*)(vp + 16), 0,1,2,3,4,5,6,7,8,9,10,11,12,13,14,15);
        O[ct] = __builtin_amdgcn_wmma_f32_16x16x32_f16(false, ap, false, bv, (short)0, O[ct], false, false);
      }
    }
  }

  // ---- epilogue: normalize by row sums, write val, accumulate IN stats ----
  float s1 = 0.f, s2 = 0.f;
  float outv[4][8];
  #pragma unroll
  for (int r = 0; r < 8; ++r) {
    float inv = 1.f / lsum[r];
    #pragma unroll
    for (int ct = 0; ct < 4; ++ct) {
      float o = O[ct][r] * inv;
      outv[ct][r] = o;
      s1 += o; s2 += o * o;
    }
  }
  const int rowbase = strip + koff8;
  #pragma unroll
  for (int ct = 0; ct < 4; ++ct)
    #pragma unroll
    for (int r = 0; r < 8; ++r)
      val[((size_t)b * H_ + rowbase + r) * C_ + ct * 16 + nlo] = outv[ct][r];

  redS[tid] = s1; redS2[tid] = s2;
  __syncthreads();
  for (int off = 128; off > 0; off >>= 1) {
    if (tid < off) { redS[tid] += redS[tid + off]; redS2[tid] += redS2[tid + off]; }
    __syncthreads();
  }
  if (tid == 0) {
    atomicAdd(&sums[b * 2 + 0], redS[0]);
    atomicAdd(&sums[b * 2 + 1], redS2[0]);
  }
}

// ---------------------------------------------------------------------------
// Kernel 6: InstanceNorm over [H,C] per batch
// ---------------------------------------------------------------------------
__global__ void norm_kernel(const float* __restrict__ val, const float* __restrict__ sums,
                            float* __restrict__ out) {
  int t = blockIdx.x * 256 + threadIdx.x;  // 2 * 8192 * 64
  int b = t / (H_ * C_);
  const float n = (float)(H_ * C_);
  float mean = sums[b * 2 + 0] / n;
  float var  = sums[b * 2 + 1] / n - mean * mean;
  out[t] = (val[t] - mean) * rsqrtf(var + 1e-5f);
}

// ---------------------------------------------------------------------------
extern "C" void kernel_launch(void* const* d_in, const int* in_sizes, int n_in,
                              void* d_out, int out_size, void* d_ws, size_t ws_size,
                              hipStream_t stream) {
  const float* q  = (const float*)d_in[0];
  const float* k  = (const float*)d_in[1];
  const float* v  = (const float*)d_in[2];
  const float* gu = (const float*)d_in[3];
  float* out = (float*)d_out;
  (void)in_sizes; (void)n_in; (void)out_size; (void)ws_size;

  char* ws = (char*)d_ws;
  size_t off = 0;
  auto alloc = [&](size_t bytes) -> char* {
    char* p = ws + off;
    off = (off + bytes + 255) & ~(size_t)255;
    return p;
  };
  float*     sq   = (float*)alloc((size_t)B_ * NB_ * HB_ * 4);     // 56 KB
  float*     sk   = (float*)alloc((size_t)B_ * NB_ * HB_ * 4);     // 56 KB
  float*     perm = (float*)alloc((size_t)B_ * 49 * 4);
  float*     sums = (float*)alloc(4 * 4);
  _Float16*  ksh  = (_Float16*)alloc((size_t)B_ * L_ * C_ * 2);    // 1 MB
  _Float16*  vth  = (_Float16*)alloc((size_t)B_ * C_ * L_ * 2);    // 1 MB
  _Float16*  qhh  = (_Float16*)alloc((size_t)B_ * H_ * C_ * 2);    // 2 MB
  float*     valb = (float*)alloc((size_t)B_ * H_ * C_ * 4);       // 4 MB

  hipLaunchKernelGGL(means_kernel,   dim3(56),      dim3(256), 0, stream, q, k, sq, sk, sums);
  hipLaunchKernelGGL(perm_kernel,    dim3(2),       dim3(256), 0, stream, sq, sk, gu, perm);
  hipLaunchKernelGGL(sortcut_kernel, dim3(512),     dim3(256), 0, stream, k, v, perm, ksh, vth);
  hipLaunchKernelGGL(qtrans_kernel,  dim3(4096),    dim3(256), 0, stream, q, qhh);
  hipLaunchKernelGGL(attn_kernel,    dim3(64, 2),   dim3(256), 0, stream, qhh, ksh, vth, valb, sums);
  hipLaunchKernelGGL(norm_kernel,    dim3(4096),    dim3(256), 0, stream, valb, sums, out);
}
